// TopologyFeatures_50457275793841
// MI455X (gfx1250) — matrix operand order repaired and checked by reading, not compiled, not run on hardware
//
#include <hip/hip_runtime.h>
#include <hip/hip_bf16.h>

typedef __attribute__((ext_vector_type(16))) _Float16 v16h;
typedef __attribute__((ext_vector_type(8)))  _Float16 v8h;
typedef __attribute__((ext_vector_type(4)))  _Float16 v4h;
typedef __attribute__((ext_vector_type(8)))  float    v8f;

#define BATCH 8192
#define FEAT  512
#define TILE  64
#define NB    (BATCH / TILE)   // 128

// ---------------------------------------------------------------------------
// Kernel 1: per-row mean/std (ddof=1) normalization, fp32 -> fp16.
// One wave (32 lanes) per row; each lane handles 4 float4 (16 elements).
// Also zeroes the global pair counter (runs before gram_count in stream order).
// ---------------------------------------------------------------------------
__global__ void topo_normalize_kernel(const float* __restrict__ x,
                                      _Float16* __restrict__ xn,
                                      unsigned* __restrict__ counter) {
  if (blockIdx.x == 0 && threadIdx.x == 0) *counter = 0u;
  const int row = blockIdx.x;
  const int t   = threadIdx.x;
  const float* xr = x + (size_t)row * FEAT;

  float4 v[4];
  float s = 0.f, q = 0.f;
#pragma unroll
  for (int i = 0; i < 4; ++i) {
    v[i] = *(const float4*)(xr + i * 128 + t * 4);
    s += v[i].x + v[i].y + v[i].z + v[i].w;
    q += v[i].x * v[i].x + v[i].y * v[i].y + v[i].z * v[i].z + v[i].w * v[i].w;
  }
#pragma unroll
  for (int off = 16; off > 0; off >>= 1) {
    s += __shfl_xor(s, off, 32);
    q += __shfl_xor(q, off, 32);
  }
  const float mean = s * (1.0f / FEAT);
  float var = (q - s * mean) * (1.0f / (FEAT - 1));   // unbiased, ddof=1
  var = var < 0.f ? 0.f : var;
  const float inv = 1.0f / (sqrtf(var) + 1e-8f);

  _Float16* yr = xn + (size_t)row * FEAT;
#pragma unroll
  for (int i = 0; i < 4; ++i) {
    v4h h;
    h.x = (_Float16)((v[i].x - mean) * inv);
    h.y = (_Float16)((v[i].y - mean) * inv);
    h.z = (_Float16)((v[i].z - mean) * inv);
    h.w = (_Float16)((v[i].w - mean) * inv);
    *(v4h*)(yr + i * 128 + t * 4) = h;
  }
}

// ---------------------------------------------------------------------------
// Kernel 2: Gram matrix tile + threshold count via WMMA fp16->fp32.
// One wave per 64x64 output tile, upper-triangular blocks only (matrix is
// bitwise symmetric); off-diagonal tiles counted twice.
//
// Fragment layouts per CDNA5 ISA (CDNA5 ISA, wave32):
//  A (16x32 f16): lanes 0-15 = row M, K = {kc..kc+7, kc+16..kc+23};
//                 lanes 16-31 = row M, K = {kc+8..kc+15, kc+24..kc+31}.
//  B (32x16 f16): lanes 0-15 = col N, K = {kc..kc+7, kc+16..kc+23};
//                 lanes 16-31 = col N, K = {kc+8..kc+15, kc+24..kc+31}.
// Both fragments come straight from the row-major fp16 x_norm.
//
// __launch_bounds__(32, 1): one wave per block, allow the full VGPR budget so
// the 16 f32 accumulators (128 VGPRs) stay resident (no scratch spills).
// ---------------------------------------------------------------------------
__global__ __launch_bounds__(32, 1)
void topo_gram_count_kernel(const _Float16* __restrict__ xn,
                            unsigned* __restrict__ counter) {
  const int bi = blockIdx.y;   // row block
  const int bj = blockIdx.x;   // col block
  if (bj < bi) return;         // symmetric: upper triangle only
  const int lane = threadIdx.x;
  const int la   = lane & 15;
  const int gA   = (lane < 16) ? 0 : 8;
  const int gB   = (lane < 16) ? 0 : 8;

  // Precomputed 32-bit element offsets (base pointer stays uniform -> SADDR).
  int offA[4], offB[4];
#pragma unroll
  for (int t = 0; t < 4; ++t) {
    offA[t] = (bi * TILE + t * 16 + la) * FEAT + gA;
    offB[t] = (bj * TILE + t * 16 + la) * FEAT + gB;
  }

  v8f c[4][4] = {};

  for (int kc = 0; kc < FEAT; kc += 32) {
    v16h a[4];
#pragma unroll
    for (int t = 0; t < 4; ++t) {
      const _Float16* pa = xn + offA[t] + kc;
      union { v16h v; v8h h[2]; } u;
      u.h[0] = *(const v8h*)(pa);
      u.h[1] = *(const v8h*)(pa + 16);
      a[t] = u.v;
    }
#pragma unroll
    for (int n = 0; n < 4; ++n) {
      union { v16h v; v8h h[2]; } ub;
      ub.h[0] = *(const v8h*)(xn + offB[n] + kc);
      ub.h[1] = *(const v8h*)(xn + offB[n] + kc + 16);
      const v16h b = ub.v;
#pragma unroll
      for (int m = 0; m < 4; ++m)
        c[m][n] = __builtin_amdgcn_wmma_f32_16x16x32_f16(
            false, a[m], false, b, (short)0, c[m][n], false, false);
    }
  }

  // corr = c / 512 > 0.5  <=>  c > 256 (exact power-of-two scaling)
  unsigned cnt = 0;
#pragma unroll
  for (int m = 0; m < 4; ++m)
#pragma unroll
    for (int n = 0; n < 4; ++n)
#pragma unroll
      for (int e = 0; e < 8; ++e)
        cnt += (c[m][n][e] > 256.0f) ? 1u : 0u;

#pragma unroll
  for (int off = 16; off > 0; off >>= 1) cnt += __shfl_xor(cnt, off, 32);

  if (lane == 0) {
    const unsigned w = (bi == bj) ? 1u : 2u;
    atomicAdd(counter, cnt * w);
  }
}

// ---------------------------------------------------------------------------
// Kernel 3: out[i,d] = t0*W[d,0] + t1*W[d,1] + t2*W[d,2] + b[d]
// with t0 = t1 = S/8192 (degree mean == clustering mean by symmetry),
// t2 = S/8192^2 (density). Every row identical; pure 16 MB store.
// ---------------------------------------------------------------------------
__global__ void topo_finalize_kernel(const unsigned* __restrict__ counter,
                                     const float* __restrict__ W,
                                     const float* __restrict__ b,
                                     float* __restrict__ out) {
  const int idx = blockIdx.x * blockDim.x + threadIdx.x;
  const float S  = (float)(*counter);
  const float t01 = S * (1.0f / 8192.0f);
  const float t2  = S * (1.0f / (8192.0f * 8192.0f));
  const int d = idx & (FEAT - 1);
  out[idx] = t01 * W[d * 3 + 0] + t01 * W[d * 3 + 1] + t2 * W[d * 3 + 2] + b[d];
}

extern "C" void kernel_launch(void* const* d_in, const int* in_sizes, int n_in,
                              void* d_out, int out_size, void* d_ws, size_t ws_size,
                              hipStream_t stream) {
  (void)in_sizes; (void)n_in; (void)out_size; (void)ws_size;
  const float* x = (const float*)d_in[0];
  const float* W = (const float*)d_in[1];
  const float* b = (const float*)d_in[2];
  float* out = (float*)d_out;

  unsigned* counter = (unsigned*)d_ws;
  _Float16* xn = (_Float16*)((char*)d_ws + 256);   // 8192*512*2 = 8 MB

  topo_normalize_kernel<<<BATCH, 32, 0, stream>>>(x, xn, counter);

  dim3 grid(NB, NB);
  topo_gram_count_kernel<<<grid, 32, 0, stream>>>(xn, counter);

  topo_finalize_kernel<<<(BATCH * FEAT) / 256, 256, 0, stream>>>(counter, W, b, out);
}
